// _B_65214783423135
// MI455X (gfx1250) — matrix operand order repaired and checked by reading, not compiled, not run on hardware
//
#include <hip/hip_runtime.h>
#include <hip/hip_bf16.h>

// ---------------------------------------------------------------- constants
constexpr int Dc = 256;
constexpr int Hc = 512;
constexpr int Rc = 16;
constexpr int Sc = 16;
constexpr int Bc = 4;
constexpr int Lc = 2048;
constexpr float EPSC = 1e-5f;

constexpr size_t DL  = (size_t)Dc * Lc;        // 524288
constexpr size_t NBL = (size_t)Bc * Lc;        // 8192
constexpr size_t U   = NBL * Dc;               // 2,097,152 floats (8 MB)

// workspace layout (floats), lifetime-based aliasing
constexpr size_t OFF_Z    = 0;        // z (B,L,D)                 [live to zz gemm]
constexpr size_t OFF_VT   = 1 * U;    // vT (B,D,L)                [dead after conv_avg]
constexpr size_t OFF_CC   = 1 * U;    //   alias: cc (B,D,L)
constexpr size_t OFF_L0   = 2 * U;    // l0                        [dead after fd gemm]
constexpr size_t OFF_CTXN = 2 * U;    //   alias: ctxn (B,L,D)
constexpr size_t OFF_H0   = 3 * U;    // h0                        [dead after fe gemm]
constexpr size_t OFF_CATL = 4 * U;    // catl (B,3D,L)  3U         [dead after fd]
constexpr size_t OFF_ZZ   = 4 * U;    //   alias: zz (B,L,2H) 4U
constexpr size_t OFF_CATH = 7 * U;    // cath (B,3D,L)  3U         [dead after fe]
constexpr size_t OFF_P    = 8 * U;    //   alias: p (B,L,H) 2U
constexpr size_t OFF_CAT2 = 10 * U;   // cat2=concat(l1,h1) 2U     [dead after ff1+rr]
constexpr size_t OFF_Q    = 10 * U;   //   alias: q (B,L,H) 2U
constexpr size_t OFF_RR   = 12 * U;   // rr (B,D,L) 1U
constexpr size_t OFF_T    = 13 * U;   // t (B,2D,L) 2U             [dead after ff2]
constexpr size_t OFF_UT   = 13 * U;   //   alias: uT (B,H,L) 2U
constexpr size_t OFF_VG   = 15 * U;   // vg (B,L,H) 2U
constexpr size_t OFF_URM  = 17 * U;   // u (B,L,H) 2U
constexpr size_t OFF_DT   = 19 * U;   // dt (B,L,H) 2U
constexpr size_t OFF_Y    = 21 * U;   // y (B,L,H) 2U
constexpr size_t OFF_ZC   = 23 * U;   // zc (B,L,48)
constexpr size_t OFF_WB   = 23 * U + NBL * 48;   // bf16 weight cache (as float slots)

// bf16 weight cache element offsets (in __bf16 elements, 32B-aligned each)
constexpr size_t WB_FD  = 0;                         // 256*768
constexpr size_t WB_FE  = WB_FD  + 256 * 768;
constexpr size_t WB_FF1 = WB_FE  + 256 * 768;        // 512*512
constexpr size_t WB_FF2 = WB_FF1 + 512 * 512;        // 256*512
constexpr size_t WB_XA  = WB_FF2 + 256 * 512;        // 1024*256
constexpr size_t WB_GB  = WB_XA  + 1024 * 256;       // 512*256
constexpr size_t WB_GC  = WB_GB  + 512 * 256;        // 512*256
constexpr size_t WB_XC  = WB_GC  + 512 * 256;        // 48*512
constexpr size_t WB_XE  = WB_XC  + 48 * 512;         // 512*16
constexpr size_t WB_XI  = WB_XE  + 512 * 16;         // 256*512

// ---------------------------------------------------------------- WMMA types
typedef __attribute__((ext_vector_type(16))) __bf16 v16bf;
typedef __attribute__((ext_vector_type(8)))  __bf16 v8bf;
typedef __attribute__((ext_vector_type(8)))  float  v8f;

// hardware f32->bf16 convert (backend emits v_cvt_pk_bf16_f32 pairs)
__device__ __forceinline__ __bf16 bf16c(float f) { return (__bf16)f; }

__device__ __forceinline__ v8f wmma_bf16(v16bf a, v16bf b, v8f c) {
  return __builtin_amdgcn_wmma_f32_16x16x32_bf16(
      /*neg_a=*/false, a, /*neg_b=*/false, b,
      /*c_mod=*/(short)0, c, /*reuse_a=*/false, /*reuse_b=*/false);
}

__device__ __forceinline__ float sigm(float x)  { return 1.0f / (1.0f + expf(-x)); }
__device__ __forceinline__ float siluf(float x) { return x * sigm(x); }
__device__ __forceinline__ float softplusf(float x) { return (x > 20.f) ? x : log1pf(expf(x)); }

// ---- fragment loaders ------------------------------------------------------
// A fragment from f32 activations: 16x32 bf16, lane = m + 16*half;
// elem j -> K=k0+half*16+j (contiguous)
__device__ __forceinline__ v16bf load_afrag(const float* __restrict__ Arow,
                                            int k0, int half, int K) {
  v16bf a;
  const int k1 = k0 + half * 16;
  const int k2 = k1 + 8;
  if (k0 + 32 <= K) {
    const float4 p0 = *(const float4*)(Arow + k1);
    const float4 p1 = *(const float4*)(Arow + k1 + 4);
    const float4 p2 = *(const float4*)(Arow + k2);
    const float4 p3 = *(const float4*)(Arow + k2 + 4);
    a[0]=bf16c(p0.x);  a[1]=bf16c(p0.y);  a[2]=bf16c(p0.z);  a[3]=bf16c(p0.w);
    a[4]=bf16c(p1.x);  a[5]=bf16c(p1.y);  a[6]=bf16c(p1.z);  a[7]=bf16c(p1.w);
    a[8]=bf16c(p2.x);  a[9]=bf16c(p2.y);  a[10]=bf16c(p2.z); a[11]=bf16c(p2.w);
    a[12]=bf16c(p3.x); a[13]=bf16c(p3.y); a[14]=bf16c(p3.z); a[15]=bf16c(p3.w);
  } else {
#pragma unroll
    for (int j = 0; j < 8; ++j) a[j]     = bf16c((k1 + j < K) ? Arow[k1 + j] : 0.0f);
#pragma unroll
    for (int j = 0; j < 8; ++j) a[8 + j] = bf16c((k2 + j < K) ? Arow[k2 + j] : 0.0f);
  }
  return a;
}

// A fragment directly from pre-converted bf16 weights (K multiple of 32)
__device__ __forceinline__ v16bf load_afrag_w(const __bf16* __restrict__ Arow,
                                              int k0, int half) {
  const int k1 = k0 + half * 16;
  const v8bf lo = *(const v8bf*)(Arow + k1);        // 16B
  const v8bf hi = *(const v8bf*)(Arow + k1 + 8);    // 16B
  return __builtin_shufflevector(lo, hi, 0, 1, 2, 3, 4, 5, 6, 7,
                                 8, 9, 10, 11, 12, 13, 14, 15);
}

// B fragment from pre-converted bf16 weight row (W (N,K) row-major):
// lane = n + 16*half, elem j -> K = k0 + half*16 + j (contiguous)
__device__ __forceinline__ v16bf load_bfrag_w(const __bf16* __restrict__ Wrow,
                                              int k0, int half, int K) {
  const int kb = k0 + half * 16;
  if (kb + 16 <= K) {
    return *(const v16bf*)(Wrow + kb);              // 32B
  }
  v16bf b;
#pragma unroll
  for (int j = 0; j < 16; ++j) b[j] = (kb + j < K) ? Wrow[kb + j] : (__bf16)0.0f;
  return b;
}

// B fragment from (K x Lc) f32 activation slab, stride Lc is compile-time so
// the 16 loads become immediate-offset global_load_b32 off one base address.
__device__ __forceinline__ v16bf load_bfrag_x(const float* __restrict__ Xcol,
                                              int k0, int half) {
  const float* Xk = Xcol + (size_t)(k0 + half * 16) * Lc;
  v16bf b;
#pragma unroll
  for (int j = 0; j < 16; ++j) b[j] = bf16c(Xk[(size_t)j * Lc]);
  return b;
}

// ---------------------------------------------------------------- GEMM (row-major act)
// C[gm, gn] = sum_k A[gm, acol0+k] * W[gn, k]    (A: M x lda f32, W: N x ldw bf16)
enum { RM_STORE = 0, RM_BIAS = 1, RM_BIAS_SIG = 2, RM_BIAS_SP = 3, RM_BIAS_RES = 4 };

template <int EPI>
__device__ __forceinline__ void store_rm(float* __restrict__ C, int ldc,
                                         const float* __restrict__ bias,
                                         const float* __restrict__ res, int ldr,
                                         v8f acc, int ms, int gn, int half) {
  float bv = 0.0f;
  if (EPI != RM_STORE) bv = bias[gn];
#pragma unroll
  for (int vi = 0; vi < 8; ++vi) {
    const int gm = ms + vi + 8 * half;
    float v = acc[vi] + bv;
    if (EPI == RM_BIAS_SIG) v = sigm(v);
    if (EPI == RM_BIAS_SP)  v = softplusf(v);
    if (EPI == RM_BIAS_RES) v += res[(size_t)gm * ldr + gn];
    C[(size_t)gm * ldc + gn] = v;
  }
}

template <int EPI>
__global__ __launch_bounds__(256) void gemm_rm_kernel(
    const float* __restrict__ A, int lda, int acol0,
    const __bf16* __restrict__ W, int ldw,
    float* __restrict__ C, int ldc,
    const float* __restrict__ bias,
    const float* __restrict__ res, int ldr,
    int M, int N, int K) {
  const int wave = blockIdx.x * (blockDim.x >> 5) + (threadIdx.x >> 5);
  const int tilesN = N >> 4;
  const int tilesM = M >> 6;
  if (wave >= tilesM * tilesN) return;           // wave-uniform: EXEC stays full
  const int mt = wave / tilesN, nt = wave % tilesN;
  const int m0 = mt << 6, n0 = nt << 4;
  const int lane = threadIdx.x & 31;
  const int half = lane >> 4, r = lane & 15;

  v8f acc0 = {}, acc1 = {}, acc2 = {}, acc3 = {};
  const float* A0 = A + (size_t)(m0 + r) * lda + acol0;
  const float* A1 = A0 + (size_t)16 * lda;
  const float* A2 = A0 + (size_t)32 * lda;
  const float* A3 = A0 + (size_t)48 * lda;
  const __bf16* Wr = W + (size_t)(n0 + r) * ldw;
  for (int k0 = 0; k0 < K; k0 += 32) {
    const v16bf bf = load_bfrag_w(Wr, k0, half, K);
    acc0 = wmma_bf16(load_afrag(A0, k0, half, K), bf, acc0);
    acc1 = wmma_bf16(load_afrag(A1, k0, half, K), bf, acc1);
    acc2 = wmma_bf16(load_afrag(A2, k0, half, K), bf, acc2);
    acc3 = wmma_bf16(load_afrag(A3, k0, half, K), bf, acc3);
  }
  const int gn = n0 + r;
  store_rm<EPI>(C, ldc, bias, res, ldr, acc0, m0,      gn, half);
  store_rm<EPI>(C, ldc, bias, res, ldr, acc1, m0 + 16, gn, half);
  store_rm<EPI>(C, ldc, bias, res, ldr, acc2, m0 + 32, gn, half);
  store_rm<EPI>(C, ldc, bias, res, ldr, acc3, m0 + 48, gn, half);
}

// ---------------------------------------------------------------- GEMM (channel-major)
// C[b, crow0+gm, gn] = sum_k A[gm,k] * X[b, k, gn]  (A: Mo x K bf16 weights, X: (B,K,Lc) f32)
enum { CM_BN_SILU = 0, CM_BN_SILU_RES = 1, CM_RES = 2 };

template <int EPI>
__device__ __forceinline__ void store_cm(float* __restrict__ Cb, int crow0,
                                         const float* __restrict__ g,
                                         const float* __restrict__ bb,
                                         const float* __restrict__ resb,
                                         v8f acc, int ms, int gn, int half, float rs) {
#pragma unroll
  for (int vi = 0; vi < 8; ++vi) {
    const int gm = ms + vi + 8 * half;
    float v = acc[vi];
    if (EPI == CM_BN_SILU || EPI == CM_BN_SILU_RES) {
      v = siluf(v * rs * g[gm] + bb[gm]);
    }
    if (EPI == CM_BN_SILU_RES || EPI == CM_RES) {
      v += resb[(size_t)gm * Lc + gn];
    }
    Cb[(size_t)(crow0 + gm) * Lc + gn] = v;
  }
}

template <int EPI>
__global__ __launch_bounds__(256) void gemm_cm_kernel(
    const __bf16* __restrict__ A,                // weights (Mo x K) row-major bf16
    const float* __restrict__ X, size_t sXB,
    float* __restrict__ C, size_t sCB, int crow0,
    const float* __restrict__ g, const float* __restrict__ bb,
    const float* __restrict__ res, size_t sRB,
    int Mo, int K, int Bn) {
  const int wave = blockIdx.x * (blockDim.x >> 5) + (threadIdx.x >> 5);
  const int tilesN = Lc >> 4;
  const int tilesM = Mo >> 6;
  const int perB = tilesM * tilesN;
  if (wave >= perB * Bn) return;                 // wave-uniform
  const int b = wave / perB;
  const int t = wave % perB;
  const int mt = t / tilesN, nt = t % tilesN;
  const int m0 = mt << 6, n0 = nt << 4;
  const int lane = threadIdx.x & 31;
  const int half = lane >> 4, r = lane & 15;

  v8f acc0 = {}, acc1 = {}, acc2 = {}, acc3 = {};
  const __bf16* A0 = A + (size_t)(m0 + r) * K;
  const __bf16* A1 = A0 + (size_t)16 * K;
  const __bf16* A2 = A0 + (size_t)32 * K;
  const __bf16* A3 = A0 + (size_t)48 * K;
  const float* Xb = X + (size_t)b * sXB + (n0 + r);
  for (int k0 = 0; k0 < K; k0 += 32) {
    const v16bf bf = load_bfrag_x(Xb, k0, half);
    acc0 = wmma_bf16(load_afrag_w(A0, k0, half), bf, acc0);
    acc1 = wmma_bf16(load_afrag_w(A1, k0, half), bf, acc1);
    acc2 = wmma_bf16(load_afrag_w(A2, k0, half), bf, acc2);
    acc3 = wmma_bf16(load_afrag_w(A3, k0, half), bf, acc3);
  }
  const float rs = rsqrtf(1.0f + EPSC);
  float* Cb = C + (size_t)b * sCB;
  const float* resb = (EPI == CM_BN_SILU) ? nullptr : (res + (size_t)b * sRB);
  const int gn = n0 + r;
  store_cm<EPI>(Cb, crow0, g, bb, resb, acc0, m0,      gn, half, rs);
  store_cm<EPI>(Cb, crow0, g, bb, resb, acc1, m0 + 16, gn, half, rs);
  store_cm<EPI>(Cb, crow0, g, bb, resb, acc2, m0 + 32, gn, half, rs);
  store_cm<EPI>(Cb, crow0, g, bb, resb, acc3, m0 + 48, gn, half, rs);
}

// ---------------------------------------------------------------- weight cast
__global__ __launch_bounds__(256) void f2bf_kernel(const float* __restrict__ src,
                                                   __bf16* __restrict__ dst, int n) {
  const int i = blockIdx.x * blockDim.x + threadIdx.x;
  if (i < n) dst[i] = (__bf16)src[i];
}

// ---------------------------------------------------------------- LayerNorm kernels
__device__ __forceinline__ float block_sum_256(float v, float* s, int d) {
  s[d] = v; __syncthreads();
#pragma unroll
  for (int o = 128; o > 0; o >>= 1) {
    if (d < o) s[d] += s[d + o];
    __syncthreads();
  }
  const float r = s[0];
  __syncthreads();
  return r;
}

__global__ __launch_bounds__(256) void ln1_kernel(
    const float* __restrict__ x, const float* __restrict__ g, const float* __restrict__ b,
    float* __restrict__ z, float* __restrict__ vT) {
  __shared__ float s[256];
  const int row = blockIdx.x;          // b*L + l
  const int d = threadIdx.x;
  const int bi = row / Lc, li = row % Lc;
  const float v = x[(size_t)row * Dc + d];
  const float m = block_sum_256(v, s, d) * (1.0f / Dc);
  const float c = v - m;
  const float var = block_sum_256(c * c, s, d) * (1.0f / Dc);
  const float zn = c * rsqrtf(var + EPSC) * g[d] + b[d];
  z[(size_t)row * Dc + d] = zn;
  vT[((size_t)bi * Dc + d) * Lc + li] = zn;
}

__global__ __launch_bounds__(256) void ln2_kernel(
    const float* __restrict__ cc,
    const float* __restrict__ fg, const float* __restrict__ fb,
    const float* __restrict__ gg, const float* __restrict__ gb,
    float* __restrict__ ctxn) {
  __shared__ float s[256];
  const int row = blockIdx.x;
  const int d = threadIdx.x;
  const int bi = row / Lc, li = row % Lc;
  const float v = cc[((size_t)bi * Dc + d) * Lc + li];
  const float m = block_sum_256(v, s, d) * (1.0f / Dc);
  const float c = v - m;
  const float var = block_sum_256(c * c, s, d) * (1.0f / Dc);
  const float ctx = c * rsqrtf(var + EPSC) * fg[d] + fb[d];
  const float m2 = block_sum_256(ctx, s, d) * (1.0f / Dc);
  const float c2 = ctx - m2;
  const float var2 = block_sum_256(c2 * c2, s, d) * (1.0f / Dc);
  ctxn[(size_t)row * Dc + d] = c2 * rsqrtf(var2 + EPSC) * gg[d] + gb[d];
}

// ---------------------------------------------------------------- depthwise convs
__device__ __forceinline__ float dwconv(const float* __restrict__ row,
                                        const float* __restrict__ w,
                                        int k, int pad, int li) {
  float sacc = 0.0f;
  for (int j = 0; j < k; ++j) {
    const int t = li + j - pad;
    if (t >= 0 && t < Lc) sacc += row[t] * w[j];
  }
  return sacc;
}

__global__ __launch_bounds__(256) void conv_avg_kernel(
    const float* __restrict__ vT, const float* __restrict__ favg,
    float* __restrict__ l0, float* __restrict__ h0) {
  const size_t idx = (size_t)blockIdx.x * blockDim.x + threadIdx.x;
  if (idx >= (size_t)Bc * Dc * Lc) return;
  const int li = (int)(idx % Lc);
  const size_t bd = idx / Lc;
  const int d = (int)(bd % Dc);
  const float* row = vT + bd * Lc;
  const float s = dwconv(row, favg + d * 9, 9, 4, li);
  l0[idx] = s;
  h0[idx] = row[li] - s;
}

__global__ __launch_bounds__(256) void conv_multi_kernel(
    const float* __restrict__ l0, const float* __restrict__ h0,
    const float* __restrict__ lk5, const float* __restrict__ lk9, const float* __restrict__ lk13,
    const float* __restrict__ hk3, const float* __restrict__ hk5, const float* __restrict__ hk7,
    float* __restrict__ catl, float* __restrict__ cath) {
  const size_t idx = (size_t)blockIdx.x * blockDim.x + threadIdx.x;
  if (idx >= (size_t)Bc * Dc * Lc) return;
  const int li = (int)(idx % Lc);
  const size_t bd = idx / Lc;
  const int d = (int)(bd % Dc);
  const int b = (int)(bd / Dc);
  const float* lr = l0 + bd * Lc;
  const float* hr = h0 + bd * Lc;
  const size_t base = (size_t)b * 3 * DL;
  catl[base + (size_t)(0 * Dc + d) * Lc + li] = dwconv(lr, lk5  + d * 5,  5,  2, li);
  catl[base + (size_t)(1 * Dc + d) * Lc + li] = dwconv(lr, lk9  + d * 9,  9,  4, li);
  catl[base + (size_t)(2 * Dc + d) * Lc + li] = dwconv(lr, lk13 + d * 13, 13, 6, li);
  cath[base + (size_t)(0 * Dc + d) * Lc + li] = dwconv(hr, hk3  + d * 3,  3,  1, li);
  cath[base + (size_t)(1 * Dc + d) * Lc + li] = dwconv(hr, hk5  + d * 5,  5,  2, li);
  cath[base + (size_t)(2 * Dc + d) * Lc + li] = dwconv(hr, hk7  + d * 7,  7,  3, li);
}

__global__ __launch_bounds__(256) void rr_kernel(const float* __restrict__ cat2,
                                                 float* __restrict__ rr) {
  const size_t idx = (size_t)blockIdx.x * blockDim.x + threadIdx.x;
  if (idx >= (size_t)Bc * Dc * Lc) return;
  const size_t b = idx / DL;
  const size_t rem = idx % DL;
  const size_t base = b * 2 * DL;
  rr[idx] = 0.5f * (cat2[base + rem] + cat2[base + DL + rem]);
}

// u = zz[:, :H] * (1 + p);  vg = zz[:, H:] + q ;  uT transposed for conv
__global__ __launch_bounds__(256) void gate_kernel(
    const float* __restrict__ zz, const float* __restrict__ p, const float* __restrict__ q,
    float* __restrict__ uT, float* __restrict__ vg) {
  const size_t idx = (size_t)blockIdx.x * blockDim.x + threadIdx.x;
  if (idx >= NBL * Hc) return;
  const int h = (int)(idx % Hc);
  const size_t bl = idx / Hc;
  const int b = (int)(bl / Lc), l = (int)(bl % Lc);
  const float u0 = zz[bl * (2 * Hc) + h];
  const float v0 = zz[bl * (2 * Hc) + Hc + h];
  uT[((size_t)b * Hc + h) * Lc + l] = u0 * (1.0f + p[bl * Hc + h]);
  vg[bl * Hc + h] = v0 + q[bl * Hc + h];
}

// causal k=3 conv (pad 2, keep first L) + bias + silu, write back row-major (B,L,H)
__global__ __launch_bounds__(256) void convxb_kernel(
    const float* __restrict__ uT, const float* __restrict__ xbw, const float* __restrict__ xbb,
    float* __restrict__ u_rm) {
  const size_t idx = (size_t)blockIdx.x * blockDim.x + threadIdx.x;
  if (idx >= (size_t)Bc * Hc * Lc) return;
  const int li = (int)(idx % Lc);
  const size_t bh = idx / Lc;
  const int h = (int)(bh % Hc);
  const int b = (int)(bh / Hc);
  const float* row = uT + bh * Lc;
  const float* w = xbw + h * 3;
  float acc = 0.0f;
#pragma unroll
  for (int j = 0; j < 3; ++j) {
    const int t = li + j - 2;
    if (t >= 0) acc += row[t] * w[j];
  }
  acc += xbb[h];
  u_rm[((size_t)b * Lc + li) * Hc + h] = siluf(acc);
}

// ---------------------------------------------------------------- SSM scan
__global__ __launch_bounds__(256) void scan_kernel(
    const float* __restrict__ dt, const float* __restrict__ u_rm,
    const float* __restrict__ zc, const float* __restrict__ vg,
    const float* __restrict__ xf, const float* __restrict__ xg,
    float* __restrict__ y) {
  const int wave = blockIdx.x * (blockDim.x >> 5) + (threadIdx.x >> 5);
  if (wave >= (Bc * Hc) / 2) return;
  const int lane = threadIdx.x & 31;
  const int s = lane & 15;
  const int b = wave / (Hc / 2);
  const int h = (wave % (Hc / 2)) * 2 + (lane >> 4);
  const float a_s = -expf(xf[h * Sc + s]);
  const float xgh = xg[h];
  float st = 0.0f;
  for (int l = 0; l < Lc; ++l) {
    const size_t bl = (size_t)b * Lc + l;
    const float dtv = dt[bl * Hc + h];
    const float uv  = u_rm[bl * Hc + h];
    const float bmv = zc[bl * 48 + Rc + s];
    const float cmv = zc[bl * 48 + Rc + Sc + s];
    st = expf(dtv * a_s) * st + (dtv * uv) * bmv;
    float yp = st * cmv;
    yp += __shfl_xor(yp, 8);
    yp += __shfl_xor(yp, 4);
    yp += __shfl_xor(yp, 2);
    yp += __shfl_xor(yp, 1);
    if (s == 0) {
      const float vgv = vg[bl * Hc + h];
      float yv = yp + uv * xgh;
      yv *= vgv * sigm(vgv);
      y[bl * Hc + h] = yv;
    }
  }
}

// ---------------------------------------------------------------- host launcher
static inline unsigned gblk(size_t waves) { return (unsigned)((waves + 7) / 8); }     // 8 waves/block
static inline unsigned eblk(size_t threads) { return (unsigned)((threads + 255) / 256); }

extern "C" void kernel_launch(void* const* d_in, const int* in_sizes, int n_in,
                              void* d_out, int out_size, void* d_ws, size_t ws_size,
                              hipStream_t stream) {
  const float* x      = (const float*)d_in[0];
  const float* ln1_g  = (const float*)d_in[1];
  const float* ln1_b  = (const float*)d_in[2];
  const float* favg_w = (const float*)d_in[3];
  const float* lk5_w  = (const float*)d_in[4];
  const float* lk9_w  = (const float*)d_in[5];
  const float* lk13_w = (const float*)d_in[6];
  const float* hk3_w  = (const float*)d_in[7];
  const float* hk5_w  = (const float*)d_in[8];
  const float* hk7_w  = (const float*)d_in[9];
  const float* fd_w   = (const float*)d_in[10];
  const float* fd_bn_g= (const float*)d_in[11];
  const float* fd_bn_b= (const float*)d_in[12];
  const float* fe_w   = (const float*)d_in[13];
  const float* fe_bn_g= (const float*)d_in[14];
  const float* fe_bn_b= (const float*)d_in[15];
  const float* ff1_w  = (const float*)d_in[16];
  const float* ff_bn_g= (const float*)d_in[17];
  const float* ff_bn_b= (const float*)d_in[18];
  const float* ff2_w  = (const float*)d_in[19];
  const float* fln_g  = (const float*)d_in[20];
  const float* fln_b  = (const float*)d_in[21];
  const float* xa_w   = (const float*)d_in[22];
  const float* xa_b   = (const float*)d_in[23];
  const float* xb_w   = (const float*)d_in[24];
  const float* xb_b   = (const float*)d_in[25];
  const float* xc_w   = (const float*)d_in[26];
  const float* xe_w   = (const float*)d_in[27];
  const float* xe_b   = (const float*)d_in[28];
  const float* xf     = (const float*)d_in[29];
  const float* xg     = (const float*)d_in[30];
  const float* xi_w   = (const float*)d_in[31];
  const float* xi_b   = (const float*)d_in[32];
  const float* gln_g  = (const float*)d_in[33];
  const float* gln_b  = (const float*)d_in[34];
  const float* gb_w   = (const float*)d_in[35];
  const float* gb_b   = (const float*)d_in[36];
  const float* gc_w   = (const float*)d_in[37];
  const float* gc_b   = (const float*)d_in[38];

  float* ws   = (float*)d_ws;
  float* z    = ws + OFF_Z;
  float* vT   = ws + OFF_VT;
  float* cc   = ws + OFF_CC;
  float* l0   = ws + OFF_L0;
  float* ctxn = ws + OFF_CTXN;
  float* h0   = ws + OFF_H0;
  float* catl = ws + OFF_CATL;
  float* zz   = ws + OFF_ZZ;
  float* cath = ws + OFF_CATH;
  float* pbuf = ws + OFF_P;
  float* cat2 = ws + OFF_CAT2;
  float* qbuf = ws + OFF_Q;
  float* rr   = ws + OFF_RR;
  float* tbuf = ws + OFF_T;
  float* uT   = ws + OFF_UT;
  float* vg   = ws + OFF_VG;
  float* urm  = ws + OFF_URM;
  float* dtb  = ws + OFF_DT;
  float* ybuf = ws + OFF_Y;
  float* zcb  = ws + OFF_ZC;
  __bf16* wb  = (__bf16*)(ws + OFF_WB);
  float* out  = (float*)d_out;

  const size_t BDL = (size_t)Bc * Dc * Lc;

  // 0) cast all GEMM weights to bf16 once
  {
    struct { const float* s; size_t off; int n; } wl[10] = {
      {fd_w,  WB_FD,  256 * 768}, {fe_w,  WB_FE,  256 * 768},
      {ff1_w, WB_FF1, 512 * 512}, {ff2_w, WB_FF2, 256 * 512},
      {xa_w,  WB_XA,  1024 * 256}, {gb_w, WB_GB,  512 * 256},
      {gc_w,  WB_GC,  512 * 256}, {xc_w,  WB_XC,  48 * 512},
      {xe_w,  WB_XE,  512 * 16},  {xi_w,  WB_XI,  256 * 512},
    };
    for (int i = 0; i < 10; ++i)
      f2bf_kernel<<<eblk(wl[i].n), 256, 0, stream>>>(wl[i].s, wb + wl[i].off, wl[i].n);
  }

  // 1) z = LN(x); vT = z^T
  ln1_kernel<<<(unsigned)NBL, 256, 0, stream>>>(x, ln1_g, ln1_b, z, vT);
  // 2) l0 = avg9(vT); h0 = vT - l0
  conv_avg_kernel<<<eblk(BDL), 256, 0, stream>>>(vT, favg_w, l0, h0);
  // 3) catl / cath filterbanks
  conv_multi_kernel<<<eblk(BDL), 256, 0, stream>>>(l0, h0, lk5_w, lk9_w, lk13_w,
                                                   hk3_w, hk5_w, hk7_w, catl, cath);
  // 4) l1 = silu(bn(fd_w @ catl)) + l0  -> cat2[:,0:256,:]
  {
    size_t waves = (size_t)Bc * (256 / 64) * (Lc / 16);
    gemm_cm_kernel<CM_BN_SILU_RES><<<gblk(waves), 256, 0, stream>>>(
        wb + WB_FD, catl, 3 * DL, cat2, 2 * DL, 0, fd_bn_g, fd_bn_b, l0, DL,
        256, 3 * Dc, Bc);
  }
  // 5) h1 = silu(bn(fe_w @ cath)) + h0  -> cat2[:,256:512,:]
  {
    size_t waves = (size_t)Bc * (256 / 64) * (Lc / 16);
    gemm_cm_kernel<CM_BN_SILU_RES><<<gblk(waves), 256, 0, stream>>>(
        wb + WB_FE, cath, 3 * DL, cat2, 2 * DL, Dc, fe_bn_g, fe_bn_b, h0, DL,
        256, 3 * Dc, Bc);
  }
  // 6) rr = 0.5*(l1+h1)
  rr_kernel<<<eblk(BDL), 256, 0, stream>>>(cat2, rr);
  // 7) t = silu(bn(ff1_w @ cat2))
  {
    size_t waves = (size_t)Bc * (512 / 64) * (Lc / 16);
    gemm_cm_kernel<CM_BN_SILU><<<gblk(waves), 256, 0, stream>>>(
        wb + WB_FF1, cat2, 2 * DL, tbuf, 2 * DL, 0, ff_bn_g, ff_bn_b, nullptr, 0,
        512, 2 * Dc, Bc);
  }
  // 8) cc = ff2_w @ t + rr
  {
    size_t waves = (size_t)Bc * (256 / 64) * (Lc / 16);
    gemm_cm_kernel<CM_RES><<<gblk(waves), 256, 0, stream>>>(
        wb + WB_FF2, tbuf, 2 * DL, cc, DL, 0, nullptr, nullptr, rr, DL,
        256, 2 * Dc, Bc);
  }
  // 9) ctxn = LN(LN(cc^T))
  ln2_kernel<<<(unsigned)NBL, 256, 0, stream>>>(cc, fln_g, fln_b, gln_g, gln_b, ctxn);
  // 10) zz = z @ xa_w^T + xa_b
  gemm_rm_kernel<RM_BIAS><<<gblk((NBL / 64) * (2 * Hc / 16)), 256, 0, stream>>>(
      z, Dc, 0, wb + WB_XA, Dc, zz, 2 * Hc, xa_b, nullptr, 0, (int)NBL, 2 * Hc, Dc);
  // 11) p = sigmoid(ctxn @ gb_w^T + gb_b)
  gemm_rm_kernel<RM_BIAS_SIG><<<gblk((NBL / 64) * (Hc / 16)), 256, 0, stream>>>(
      ctxn, Dc, 0, wb + WB_GB, Dc, pbuf, Hc, gb_b, nullptr, 0, (int)NBL, Hc, Dc);
  // 12) q = ctxn @ gc_w^T + gc_b
  gemm_rm_kernel<RM_BIAS><<<gblk((NBL / 64) * (Hc / 16)), 256, 0, stream>>>(
      ctxn, Dc, 0, wb + WB_GC, Dc, qbuf, Hc, gc_b, nullptr, 0, (int)NBL, Hc, Dc);
  // 13) gates: uT = u*(1+p) (transposed), vg = vg + q
  gate_kernel<<<eblk(NBL * Hc), 256, 0, stream>>>(zz, pbuf, qbuf, uT, vg);
  // 14) u = silu(causal_conv3(uT) + xb_b), row-major
  convxb_kernel<<<eblk((size_t)Bc * Hc * Lc), 256, 0, stream>>>(uT, xb_w, xb_b, urm);
  // 15) zc = u @ xc_w^T
  gemm_rm_kernel<RM_STORE><<<gblk((NBL / 64) * (48 / 16)), 256, 0, stream>>>(
      urm, Hc, 0, wb + WB_XC, Hc, zcb, 48, nullptr, nullptr, 0, (int)NBL, 48, Hc);
  // 16) dt = softplus(zc[:,:16] @ xe_w^T + xe_b)
  gemm_rm_kernel<RM_BIAS_SP><<<gblk((NBL / 64) * (Hc / 16)), 256, 0, stream>>>(
      zcb, 48, 0, wb + WB_XE, Rc, dtb, Hc, xe_b, nullptr, 0, (int)NBL, Hc, Rc);
  // 17) SSM scan + output gating -> y
  scan_kernel<<<gblk((size_t)(Bc * Hc) / 2), 256, 0, stream>>>(
      dtb, urm, zcb, vg, xf, xg, ybuf);
  // 18) out = y @ xi_w^T + xi_b + x
  gemm_rm_kernel<RM_BIAS_RES><<<gblk((NBL / 64) * (Dc / 16)), 256, 0, stream>>>(
      ybuf, Hc, 0, wb + WB_XI, Hc, out, Dc, xi_b, x, Dc, (int)NBL, Dc, Hc);
}
